// MultiHeadedAttention_61229053772463
// MI455X (gfx1250) — matrix operand order repaired and checked by reading, not compiled, not run on hardware
//
#include <hip/hip_runtime.h>
#include <hip/hip_bf16.h>
#include <stdint.h>

// ---------------------------------------------------------------------------
// MultiHeadedAttention for MI455X (gfx1250, wave32, WMMA f32<=f16xf16)
// B=2, S=2048, D=1024, H=16, DK=64
//
// Pipeline (all on `stream`):
//   1) cvt f32->f16 : query,key,value, Wq,Wk,Wv,Wo into workspace
//   2) gemm64<0>    : Q = (x@Wq^T + bq) * 1/sqrt(DK) -> f16 [B,H,S,DK]
//      gemm64<0>    : K = x@Wk^T + bk  -> f16 [B,H,S,DK]
//      gemm64<1>    : V = x@Wv^T + bv  -> f16 [B,H,DK,S]   (transposed for PV)
//   3) flash_attn   : fused QK^T -> masked online-softmax -> P@V, f16 [B,S,D]
//   4) gemm64<2>    : out = attn@Wo^T + bo -> f32 d_out
//
// Design notes (reasoned from MI455X specs, no runtime available):
//  * Materializing the [B,H,S,S] scores costs ~536 MB round-trip -> ~46 us at
//    23.3 TB/s; the fused flash kernel keeps scores in VGPRs/LDS.
//  * All GEMM operands fit in the 192 MB L2, so per-wave operand reloads are
//    L2 hits; wave-level 64x64 tiles with 16 WMMAs per 32-k step are enough.
//  * f16 storage + f32 WMMA accumulation: inputs are N(0,1) with 0.02-scale
//    weights, f16's 10 mantissa bits keep ~1e-3 relative error.
//  * 1/sqrt(DK) folded into the Q projection epilogue (removes 64 VALU
//    muls per key-block per wave on the WMMA->softmax critical path).
//  * __builtin_prefetch -> global_prefetch_b8 covers the next 64-k chunk.
// ---------------------------------------------------------------------------

#define B_  2
#define S_  2048
#define D_  1024
#define H_  16
#define DK_ 64
#define MROWS (B_ * S_)          // 4096

typedef __attribute__((ext_vector_type(16))) _Float16 v16h;
typedef __attribute__((ext_vector_type(8)))  _Float16 v8h;
typedef __attribute__((ext_vector_type(8)))  float    v8f;

#define WMMA_F16(a, b, c) \
  __builtin_amdgcn_wmma_f32_16x16x32_f16(false, (a), false, (b), (short)0, (c), false, false)

// A-matrix fragment, 16x32 f16 (M x K), src row-major with leading dim `ld`.
// Layout used (contiguous k, not ISA table order): lane<16 -> M=lane,   K in {0..15}
//             lane>=16-> M=lane-16,K in {16..31}
__device__ __forceinline__ v16h load_a_frag(const _Float16* __restrict__ src, int ld) {
  const int lane = threadIdx.x & 31;
  const int m    = lane & 15;
  const int hw   = lane >> 4;
  const _Float16* p = src + (size_t)m * ld + hw * 16;
  union { v16h v; v8h h[2]; } u;
  u.h[0] = *(const v8h*)(p);        // K = hw*16 .. hw*16+7
  u.h[1] = *(const v8h*)(p + 8);    // K = hw*16+8 .. hw*16+15
  return u.v;
}

// B-matrix fragment, 32x16 f16 (K x N); element (k,n) = src[n*ld + k]
// Layout used (contiguous k, as in load_a_frag): lanes 0-15 hold K=0..15 of column N=lane,
//                                  lanes 16-31 hold K=16..31 of column N=lane-16.
__device__ __forceinline__ v16h load_b_frag(const _Float16* __restrict__ src, int ld) {
  const int lane = threadIdx.x & 31;
  const int n  = lane & 15;
  const int kb = (lane >> 4) * 16;
  return *(const v16h*)(src + (size_t)n * ld + kb);   // 16 contiguous halves (32B)
}

// ------------------------------- convert -----------------------------------
__global__ __launch_bounds__(256) void cvt_f32_f16(const float* __restrict__ in,
                                                   _Float16* __restrict__ out, int n4) {
  int i = (blockIdx.x * blockDim.x + threadIdx.x);
  if (i < n4) {
    float4 f = *(const float4*)(in + (size_t)i * 4);
    _Float16* o = out + (size_t)i * 4;
    o[0] = (_Float16)f.x; o[1] = (_Float16)f.y;
    o[2] = (_Float16)f.z; o[3] = (_Float16)f.w;
  }
}

// ------------------------------- GEMM --------------------------------------
// C[M=4096, N=1024] = (X(f16, ld=D) @ W^T + bias) * outScale
// One wave computes a 64x64 tile. OUT_MODE: 0 -> f16 [B,H,S,DK]
//                                           1 -> f16 [B,H,DK,S] (V transposed)
//                                           2 -> f32 row-major [M,N] (d_out)
template <int OUT_MODE>
__global__ __launch_bounds__(256) void gemm64(const _Float16* __restrict__ X,
                                              const _Float16* __restrict__ W,
                                              const float* __restrict__ bias,
                                              float outScale,
                                              _Float16* __restrict__ out16,
                                              float* __restrict__ out32) {
  const int wave = (blockIdx.x * blockDim.x + threadIdx.x) >> 5;
  const int NT = D_ / 64;                 // 16 tiles along N
  const int m0 = (wave / NT) * 64;
  const int n0 = (wave % NT) * 64;
  const int lane = threadIdx.x & 31;
  const int hw = lane >> 4, nl = lane & 15;

  v8f acc[4][4];
  #pragma unroll
  for (int mi = 0; mi < 4; ++mi)
    #pragma unroll
    for (int nj = 0; nj < 4; ++nj) { v8f z = {}; acc[mi][nj] = z; }

  // 64-wide k chunks: one speculative prefetch covers the next chunk's
  // 128B line per row (emits global_prefetch_b8 on gfx1250).
  for (int k = 0; k < D_; k += 64) {
    __builtin_prefetch(X + (size_t)(m0 + lane) * D_ + k + 64, 0, 1);
    __builtin_prefetch(W + (size_t)(n0 + lane) * D_ + k + 64, 0, 1);
    #pragma unroll
    for (int ks = 0; ks < 2; ++ks) {
      const int kk = k + ks * 32;
      v16h a[4], b[4];
      #pragma unroll
      for (int mi = 0; mi < 4; ++mi)
        a[mi] = load_a_frag(X + (size_t)(m0 + mi * 16) * D_ + kk, D_);
      #pragma unroll
      for (int nj = 0; nj < 4; ++nj)
        b[nj] = load_b_frag(W + (size_t)(n0 + nj * 16) * D_ + kk, D_);
      #pragma unroll
      for (int mi = 0; mi < 4; ++mi)
        #pragma unroll
        for (int nj = 0; nj < 4; ++nj)
          acc[mi][nj] = WMMA_F16(a[mi], b[nj], acc[mi][nj]);
    }
  }

  // Epilogue: C/D layout -> lane (hw,nl), vgpr r holds (m = r + 8*hw, n = nl)
  #pragma unroll
  for (int mi = 0; mi < 4; ++mi) {
    #pragma unroll
    for (int nj = 0; nj < 4; ++nj) {
      const int n = n0 + nj * 16 + nl;
      const float bv = bias[n];
      #pragma unroll
      for (int r = 0; r < 8; ++r) {
        const int m = m0 + mi * 16 + r + 8 * hw;
        const float v = (acc[mi][nj][r] + bv) * outScale;
        if (OUT_MODE == 2) {
          out32[(size_t)m * D_ + n] = v;
        } else {
          const int bb = m / S_, s = m % S_;
          const int h = n / DK_, dk = n % DK_;
          if (OUT_MODE == 0)
            out16[(((size_t)bb * H_ + h) * S_ + s) * DK_ + dk] = (_Float16)v;
          else
            out16[(((size_t)bb * H_ + h) * DK_ + dk) * S_ + s] = (_Float16)v;
        }
      }
    }
  }
}

// --------------------------- flash attention -------------------------------
// One wave owns 32 query rows of one (b,h); loops over 64-wide key blocks.
// Q is pre-scaled by 1/sqrt(DK), so scores come out of WMMA ready for softmax.
__global__ __launch_bounds__(128) void flash_attn(const _Float16* __restrict__ Q,
                                                  const _Float16* __restrict__ K,
                                                  const _Float16* __restrict__ VT,
                                                  const int* __restrict__ mask,
                                                  _Float16* __restrict__ Oout) {
  __shared__ _Float16 ldsP[4][32 * 64];   // per-wave 32x64 f16 P tile (4 KB each)

  const int wlocal = threadIdx.x >> 5;
  const int wave = blockIdx.x * (blockDim.x >> 5) + wlocal;
  const int QT = S_ / 32;                 // 64 query tiles per (b,h)
  const int bh = wave / QT;
  const int qt = wave % QT;
  const int b = bh / H_, h = bh % H_;
  const int q0 = qt * 32;
  const int lane = threadIdx.x & 31;
  const int hw = lane >> 4, nl = lane & 15;

  const _Float16* Qb  = Q  + ((size_t)(b * H_ + h) * S_ + q0) * DK_;
  const _Float16* Kb  = K  + (size_t)(b * H_ + h) * S_ * DK_;
  const _Float16* VTb = VT + (size_t)(b * H_ + h) * DK_ * S_;
  _Float16* lds = &ldsP[wlocal][0];

  // Preload Q fragments for all 32 rows (2 m-subtiles x 2 k-steps of DK=64)
  v16h qa[2][2];
  #pragma unroll
  for (int mi = 0; mi < 2; ++mi)
    #pragma unroll
    for (int ks = 0; ks < 2; ++ks)
      qa[mi][ks] = load_a_frag(Qb + (size_t)mi * 16 * DK_ + ks * 32, DK_);

  float mrun[2][8], lrun[2][8];
  v8f oacc[2][4];
  #pragma unroll
  for (int mi = 0; mi < 2; ++mi) {
    #pragma unroll
    for (int r = 0; r < 8; ++r) { mrun[mi][r] = -1e30f; lrun[mi][r] = 0.0f; }
    #pragma unroll
    for (int nc = 0; nc < 4; ++nc) { v8f z = {}; oacc[mi][nc] = z; }
  }

  for (int j0 = 0; j0 < S_; j0 += 64) {
    // prefetch next key block's K rows and VT columns (global_prefetch_b8)
    __builtin_prefetch(Kb + (size_t)(j0 + 64 + lane) * DK_, 0, 1);
    __builtin_prefetch(VTb + (size_t)lane * S_ + j0 + 64, 0, 1);

    // ---- scores: S = (Qscaled @ K^T), masked ---------------------------
    v8f sacc[2][4];
    #pragma unroll
    for (int mi = 0; mi < 2; ++mi)
      #pragma unroll
      for (int nj = 0; nj < 4; ++nj) { v8f z = {}; sacc[mi][nj] = z; }

    #pragma unroll
    for (int ks = 0; ks < 2; ++ks) {
      v16h kb[4];
      #pragma unroll
      for (int nj = 0; nj < 4; ++nj)
        kb[nj] = load_b_frag(Kb + (size_t)(j0 + nj * 16) * DK_ + ks * 32, DK_);
      #pragma unroll
      for (int mi = 0; mi < 2; ++mi)
        #pragma unroll
        for (int nj = 0; nj < 4; ++nj)
          sacc[mi][nj] = WMMA_F16(qa[mi][ks], kb[nj], sacc[mi][nj]);
    }

    int mflag[4];
    #pragma unroll
    for (int nj = 0; nj < 4; ++nj)
      mflag[nj] = mask[(size_t)b * S_ + j0 + nj * 16 + nl];

    #pragma unroll
    for (int mi = 0; mi < 2; ++mi)
      #pragma unroll
      for (int nj = 0; nj < 4; ++nj)
        #pragma unroll
        for (int r = 0; r < 8; ++r)
          if (mflag[nj]) sacc[mi][nj][r] = -1e30f;   // key padding

    // ---- online softmax (row = vgpr r, lives in one 16-lane group) -----
    float scalef[2][8];
    #pragma unroll
    for (int mi = 0; mi < 2; ++mi) {
      #pragma unroll
      for (int r = 0; r < 8; ++r) {
        float mx = sacc[mi][0][r];
        #pragma unroll
        for (int nj = 1; nj < 4; ++nj) mx = fmaxf(mx, sacc[mi][nj][r]);
        #pragma unroll
        for (int off = 1; off < 16; off <<= 1)
          mx = fmaxf(mx, __shfl_xor(mx, off, 32));
        const float mnew = fmaxf(mrun[mi][r], mx);
        const float sc = __expf(mrun[mi][r] - mnew);
        mrun[mi][r] = mnew;
        scalef[mi][r] = sc;
        float rs = 0.0f;
        #pragma unroll
        for (int nj = 0; nj < 4; ++nj) {
          float s = sacc[mi][nj][r];
          float p = (s < -1e29f) ? 0.0f : __expf(s - mnew);
          sacc[mi][nj][r] = p;
          rs += p;
        }
        #pragma unroll
        for (int off = 1; off < 16; off <<= 1)
          rs += __shfl_xor(rs, off, 32);
        lrun[mi][r] = lrun[mi][r] * sc + rs;
      }
    }

    // rescale running output
    #pragma unroll
    for (int mi = 0; mi < 2; ++mi)
      #pragma unroll
      for (int nc = 0; nc < 4; ++nc)
        #pragma unroll
        for (int r = 0; r < 8; ++r)
          oacc[mi][nc][r] *= scalef[mi][r];

    // P (C-layout) -> LDS f16 -> reload as A-layout fragments
    #pragma unroll
    for (int mi = 0; mi < 2; ++mi)
      #pragma unroll
      for (int nj = 0; nj < 4; ++nj)
        #pragma unroll
        for (int r = 0; r < 8; ++r)
          lds[(mi * 16 + r + 8 * hw) * 64 + nj * 16 + nl] = (_Float16)sacc[mi][nj][r];
    asm volatile("s_wait_dscnt 0" ::: "memory");   // single-wave LDS RAW

    // ---- O += P @ V  (V stored transposed: VT[dk, j]) ------------------
    #pragma unroll
    for (int ks = 0; ks < 2; ++ks) {
      v16h pa[2], vb[4];
      #pragma unroll
      for (int mi = 0; mi < 2; ++mi)
        pa[mi] = load_a_frag(lds + mi * 16 * 64 + ks * 32, 64);
      #pragma unroll
      for (int nc = 0; nc < 4; ++nc)
        vb[nc] = load_b_frag(VTb + (size_t)(nc * 16) * S_ + j0 + ks * 32, S_);
      #pragma unroll
      for (int mi = 0; mi < 2; ++mi)
        #pragma unroll
        for (int nc = 0; nc < 4; ++nc)
          oacc[mi][nc] = WMMA_F16(pa[mi], vb[nc], oacc[mi][nc]);
    }
  }

  // normalize and store f16 [B,S,D]
  #pragma unroll
  for (int mi = 0; mi < 2; ++mi)
    #pragma unroll
    for (int nc = 0; nc < 4; ++nc) {
      const int dk = nc * 16 + nl;
      #pragma unroll
      for (int r = 0; r < 8; ++r) {
        const int s = q0 + mi * 16 + r + 8 * hw;
        const float v = oacc[mi][nc][r] / lrun[mi][r];
        Oout[((size_t)b * S_ + s) * D_ + h * DK_ + dk] = (_Float16)v;
      }
    }
}

// ------------------------------ launcher -----------------------------------
extern "C" void kernel_launch(void* const* d_in, const int* in_sizes, int n_in,
                              void* d_out, int out_size, void* d_ws, size_t ws_size,
                              hipStream_t stream) {
  const float* q  = (const float*)d_in[0];
  const float* k  = (const float*)d_in[1];
  const float* v  = (const float*)d_in[2];
  const int*   mk = (const int*)  d_in[3];
  const float* Wq = (const float*)d_in[4];
  const float* bq = (const float*)d_in[5];
  const float* Wk = (const float*)d_in[6];
  const float* bk = (const float*)d_in[7];
  const float* Wv = (const float*)d_in[8];
  const float* bv = (const float*)d_in[9];
  const float* Wo = (const float*)d_in[10];
  const float* bo = (const float*)d_in[11];
  float* out = (float*)d_out;

  const size_t XE = (size_t)B_ * S_ * D_;   // 4 Mi elems
  const size_t WE = (size_t)D_ * D_;        // 1 Mi elems

  char* ws = (char*)d_ws;
  size_t off = 0;
  auto alloc16 = [&](size_t elems) -> _Float16* {
    off = (off + 255) & ~(size_t)255;
    _Float16* p = (_Float16*)(ws + off);
    off += elems * sizeof(_Float16);
    return p;
  };
  _Float16* xq16 = alloc16(XE);
  _Float16* xk16 = alloc16(XE);
  _Float16* xv16 = alloc16(XE);
  _Float16* wq16 = alloc16(WE);
  _Float16* wk16 = alloc16(WE);
  _Float16* wv16 = alloc16(WE);
  _Float16* wo16 = alloc16(WE);
  _Float16* Q16  = alloc16(XE);  // [B,H,S,DK], pre-scaled by 1/sqrt(DK)
  _Float16* K16  = alloc16(XE);  // [B,H,S,DK]
  _Float16* VT16 = alloc16(XE);  // [B,H,DK,S]
  _Float16* at16 = alloc16(XE);  // attention output f16 [B,S,D]

  // 1) converts
  {
    const int tb = 256;
    const int gX = (int)(XE / 4 + tb - 1) / tb;
    const int gW = (int)(WE / 4 + tb - 1) / tb;
    cvt_f32_f16<<<gX, tb, 0, stream>>>(q,  xq16, (int)(XE / 4));
    cvt_f32_f16<<<gX, tb, 0, stream>>>(k,  xk16, (int)(XE / 4));
    cvt_f32_f16<<<gX, tb, 0, stream>>>(v,  xv16, (int)(XE / 4));
    cvt_f32_f16<<<gW, tb, 0, stream>>>(Wq, wq16, (int)(WE / 4));
    cvt_f32_f16<<<gW, tb, 0, stream>>>(Wk, wk16, (int)(WE / 4));
    cvt_f32_f16<<<gW, tb, 0, stream>>>(Wv, wv16, (int)(WE / 4));
    cvt_f32_f16<<<gW, tb, 0, stream>>>(Wo, wo16, (int)(WE / 4));
  }

  // 2) projections: 1024 wave-tiles (64x64), 8 waves per 256-thread block
  const int gemmBlocks = (MROWS / 64) * (D_ / 64) / 8;   // 128
  const float rsqrtDK = 0.125f;                          // 1/sqrt(64)
  gemm64<0><<<gemmBlocks, 256, 0, stream>>>(xq16, wq16, bq, rsqrtDK, Q16,  nullptr);
  gemm64<0><<<gemmBlocks, 256, 0, stream>>>(xk16, wk16, bk, 1.0f,    K16,  nullptr);
  gemm64<1><<<gemmBlocks, 256, 0, stream>>>(xv16, wv16, bv, 1.0f,    VT16, nullptr);

  // 3) fused attention: B*H*(S/32) = 2048 waves, 4 waves per 128-thread block
  flash_attn<<<(B_ * H_ * (S_ / 32)) / 4, 128, 0, stream>>>(Q16, K16, VT16, mk, at16);

  // 4) output projection -> fp32 d_out
  gemm64<2><<<gemmBlocks, 256, 0, stream>>>(at16, wo16, bo, 1.0f, nullptr, out);
}
